// ViT_12549894439283
// MI455X (gfx1250) — compile-verified, matrix operand order checked by reading, not run on hardware
//
#include <hip/hip_runtime.h>
#include <hip/hip_bf16.h>

typedef __attribute__((ext_vector_type(16))) _Float16 v16h;
typedef __attribute__((ext_vector_type(8)))  float    v8f;

#define KCH   256
#define HWDIM 64
#define HWSZ  4096      // 64*64
#define BTF   32
#define TSEQ  4
#define NANCH 128       // B * GRID_N^2 = 8*16

// ---------------- ws layout (byte offsets) ----------------
#define OFF_SRC1 0u                 // int[32*4096]
#define OFF_SRC2 524288u            // int[32*4096]
#define OFF_SRCR 1048576u           // int[32*4096]
#define OFF_Q1   1572864u           // _Float16[128*256]
#define OFF_X1   1638400u           // _Float16[128*256]
#define OFF_Y1   1703936u           // _Float16[128*256]
#define OFF_ACC  1769472u           // float[64]: [0]=ce_sum [1]=ref_loss [2..33]=frame sums

// ---------------------------------------------------------------------------
__global__ void init_kernel(float* acc) {
    int t = threadIdx.x;
    if (t < 64) acc[t] = 0.0f;
}

// ---------------------------------------------------------------------------
__device__ __forceinline__ int nearest_zeros(float x, float y) {
    float ix = (x + 1.0f) * 32.0f - 0.5f;
    float iy = (y + 1.0f) * 32.0f - 0.5f;
    int jx = (int)rintf(ix);          // round-half-to-even, matches jnp.rint
    int jy = (int)rintf(iy);
    if (jx < 0 || jx > 63 || jy < 0 || jy > 63) return -1;
    return jy * HWDIM + jx;
}

__device__ __forceinline__ int nearest_reflect(float x, float y) {
    float ix = (x + 1.0f) * 32.0f - 0.5f;
    float iy = (y + 1.0f) * 32.0f - 0.5f;
    float cx = fmodf(fabsf(ix + 0.5f), 128.0f);
    cx = ((cx >= 64.0f) ? (128.0f - cx) : cx) - 0.5f;
    float cy = fmodf(fabsf(iy + 0.5f), 128.0f);
    cy = ((cy >= 64.0f) ? (128.0f - cy) : cy) - 0.5f;
    int jx = (int)rintf(cx);
    int jy = (int)rintf(cy);
    jx = jx < 0 ? 0 : (jx > 63 ? 63 : jx);
    jy = jy < 0 ? 0 : (jy > 63 ? 63 : jy);
    return jy * HWDIM + jx;
}

__global__ __launch_bounds__(256)
void precompute_kernel(const float* __restrict__ aff1, const float* __restrict__ aff2,
                       const float* __restrict__ aff_grid,
                       int* __restrict__ src1, int* __restrict__ src2, int* __restrict__ srcR) {
    int n   = blockIdx.y;                          // frame
    int pix = blockIdx.x * 256 + threadIdx.x;      // 0..4095
    if (pix >= HWSZ) return;
    int h = pix >> 6, w = pix & 63;
    float gx = (w + 0.5f) * (2.0f / 64.0f) - 1.0f;
    float gy = (h + 0.5f) * (2.0f / 64.0f) - 1.0f;

    const float* t1 = aff1 + n * 6;
    float x1 = t1[0] * gx + t1[1] * gy + t1[2];
    float y1 = t1[3] * gx + t1[4] * gy + t1[5];
    src1[n * HWSZ + pix] = nearest_zeros(x1, y1);

    const float* t2 = aff2 + n * 6;
    float x2 = t2[0] * gx + t2[1] * gy + t2[2];
    float y2 = t2[3] * gx + t2[4] * gy + t2[5];
    src2[n * HWSZ + pix] = nearest_zeros(x2, y2);

    float ax = aff_grid[(size_t)(n * HWSZ + pix) * 2 + 0];
    float ay = aff_grid[(size_t)(n * HWSZ + pix) * 2 + 1];
    srcR[n * HWSZ + pix] = nearest_reflect(ax, ay);
}

// ---------------------------------------------------------------------------
// query1[a][c] = k1[frame = 4*(a/16), c, hw_index[a]]   (f16)
__global__ __launch_bounds__(256)
void anchors_kernel(const float* __restrict__ k1, const int* __restrict__ hw_index,
                    _Float16* __restrict__ q1) {
    int a = blockIdx.x;           // 0..127
    int c = threadIdx.x;          // 0..255
    int b = a >> 4;
    int idx = hw_index[a];
    size_t n = (size_t)b * TSEQ;
    q1[a * KCH + c] = (_Float16)k1[(n * KCH + c) * HWSZ + idx];
}

// x1[a][c] = valid1 ? k1[n=4b, c, src1[n, hw_ref[a]]] : 0, y1 analog with k2/src2
__global__ __launch_bounds__(256)
void refgather_kernel(const float* __restrict__ k1, const float* __restrict__ k2,
                      const int* __restrict__ src1, const int* __restrict__ src2,
                      const int* __restrict__ hw_ref,
                      _Float16* __restrict__ x1, _Float16* __restrict__ y1) {
    int a = blockIdx.x, c = threadIdx.x;
    int b = a >> 4;
    int n = b * TSEQ;
    int idx = hw_ref[a];
    int s1 = src1[n * HWSZ + idx];
    int s2 = src2[n * HWSZ + idx];
    size_t fb = (size_t)n * KCH * HWSZ;
    x1[a * KCH + c] = (_Float16)(s1 < 0 ? 0.0f : k1[fb + (size_t)c * HWSZ + s1]);
    y1[a * KCH + c] = (_Float16)(s2 < 0 ? 0.0f : k2[fb + (size_t)c * HWSZ + s2]);
}

// ---------------------------------------------------------------------------
// Main CE loss: per block = 64 pixels of one frame (t>=1 only).
// Waves 0-3: logits1 = sampled-k1 . q1^T (8 N-tiles each).
// Waves 4-7: logits2 only for the frame's own 16-anchor block (1 N-tile).
__global__ __launch_bounds__(256)
void temp_loss_kernel(const float* __restrict__ k1, const float* __restrict__ k2,
                      const int* __restrict__ src1, const int* __restrict__ src2,
                      const _Float16* __restrict__ q1, float* __restrict__ acc_g) {
    int bid = blockIdx.x;                // 0..1535
    int fi  = bid >> 6;                  // 0..23  (frames with t>=1)
    int mb  = bid & 63;                  // 64-pixel block within frame
    int n   = (fi / 3) * 4 + 1 + (fi % 3);
    int vid = n >> 2;
    int p0  = mb * 64;
    size_t fbase = (size_t)n * KCH * HWSZ;

    __shared__ _Float16 sA1[64 * 32];
    __shared__ _Float16 sA2[64 * 32];
    __shared__ _Float16 sB [128 * 32];
    __shared__ float    sL1[64 * 128];
    __shared__ float    sL2[64 * 16];
    __shared__ int      sS1[64], sS2[64];
    __shared__ float    sRed[256];

    int tid  = threadIdx.x;
    int lane = tid & 31;
    int wv   = tid >> 5;
    int mt   = wv & 3;
    bool isM2 = (wv >= 4);

    if (tid < 64) {
        sS1[tid] = src1[n * HWSZ + p0 + tid];
        sS2[tid] = src2[n * HWSZ + p0 + tid];
    }

    v8f acc[8];
    #pragma unroll
    for (int i = 0; i < 8; i++)
        #pragma unroll
        for (int r = 0; r < 8; r++) acc[i][r] = 0.0f;

    for (int kc = 0; kc < 8; kc++) {
        __syncthreads();
        // stage sampled A tiles (64 px x 32 k) and B tile (128 anch x 32 k)
        for (int i = tid; i < 64 * 32; i += 256) {
            int p = i >> 5, k = i & 31;
            int c = kc * 32 + k;
            int s1 = sS1[p], s2 = sS2[p];
            sA1[i] = (_Float16)(s1 < 0 ? 0.0f : k1[fbase + (size_t)c * HWSZ + s1]);
            sA2[i] = (_Float16)(s2 < 0 ? 0.0f : k2[fbase + (size_t)c * HWSZ + s2]);
        }
        for (int i = tid; i < 128 * 32; i += 256) {
            int a = i >> 5, k = i & 31;
            sB[i] = q1[a * KCH + kc * 32 + k];
        }
        __syncthreads();

        // A fragment per documented 16-bit 16x32 layout:
        //   lane l: M = l&15, element e -> K = (e<8 ? e : e+8) + 8*(l>>4)
        v16h afrag;
        {
            int m   = lane & 15;
            int khi = (lane >> 4) * 8;
            const _Float16* sA = isM2 ? sA2 : sA1;
            int prow = mt * 16 + m;
            #pragma unroll
            for (int e = 0; e < 16; e++) {
                int k = (e < 8 ? e : e + 8) + khi;
                afrag[e] = sA[prow * 32 + k];
            }
        }
        int khi = (lane >> 4) * 8;
        if (!isM2) {
            #pragma unroll
            for (int nt = 0; nt < 8; nt++) {
                v16h bfrag;
                int ncol = nt * 16 + (lane & 15);
                #pragma unroll
                for (int e = 0; e < 16; e++) {
                    int k = (e < 8 ? e : e + 8) + khi;
                    bfrag[e] = sB[ncol * 32 + k];
                }
                acc[nt] = __builtin_amdgcn_wmma_f32_16x16x32_f16(
                    false, afrag, false, bfrag, (short)0, acc[nt], false, false);
            }
        } else {
            v16h bfrag;
            int ncol = vid * 16 + (lane & 15);
            #pragma unroll
            for (int e = 0; e < 16; e++) {
                int k = (e < 8 ? e : e + 8) + khi;
                bfrag[e] = sB[ncol * 32 + k];
            }
            acc[0] = __builtin_amdgcn_wmma_f32_16x16x32_f16(
                false, afrag, false, bfrag, (short)0, acc[0], false, false);
        }
    }
    __syncthreads();

    // D layout: lane l, vgpr r -> M = r + 8*(l>>4), N = l&15
    const float inv_temp = 1.0f / 25.0f;
    int mrow0 = (lane >> 4) * 8;
    int ncl   = lane & 15;
    if (!isM2) {
        #pragma unroll
        for (int nt = 0; nt < 8; nt++)
            #pragma unroll
            for (int r = 0; r < 8; r++) {
                int p = mt * 16 + mrow0 + r;
                float v = (sS1[p] >= 0) ? acc[nt][r] * inv_temp : 0.0f;
                sL1[p * 128 + nt * 16 + ncl] = v;
            }
    } else {
        #pragma unroll
        for (int r = 0; r < 8; r++) {
            int p = mt * 16 + mrow0 + r;
            float v = (sS2[p] >= 0) ? acc[0][r] * inv_temp : 0.0f;
            sL2[p * 16 + ncl] = v;
        }
    }
    __syncthreads();

    // per-pixel: argmax over own 16-block of logits2, CE over 128 logits1
    float ce = 0.0f;
    if (tid < 64) {
        int p = tid;
        int best = 0; float bv = sL2[p * 16];
        for (int j = 1; j < 16; j++) {
            float v = sL2[p * 16 + j];
            if (v > bv) { bv = v; best = j; }     // first-max tie rule
        }
        int pc = vid * 16 + best;
        float mx = sL1[p * 128];
        for (int j = 1; j < 128; j++) mx = fmaxf(mx, sL1[p * 128 + j]);
        float se = 0.0f;
        for (int j = 0; j < 128; j++) se += expf(sL1[p * 128 + j] - mx);
        ce = (mx + logf(se)) - sL1[p * 128 + pc];
    }
    sRed[tid] = ce;
    __syncthreads();
    for (int s = 128; s > 0; s >>= 1) {
        if (tid < s) sRed[tid] += sRed[tid + s];
        __syncthreads();
    }
    if (tid == 0) atomicAdd(&acc_g[0], sRed[0]);
}

// ---------------------------------------------------------------------------
// ref loss: one workgroup, logits = x1 . y1^T / 25 (128x128x256 WMMA),
// loss = mean(lse_row - diag)
__global__ __launch_bounds__(256)
void ref_loss_kernel(const _Float16* __restrict__ x1, const _Float16* __restrict__ y1,
                     float* __restrict__ acc_g) {
    __shared__ __align__(16) unsigned char smem[128 * 128 * 4];   // 64 KB, reused
    _Float16* sX = (_Float16*)smem;            // 128*32 f16 (8 KB)
    _Float16* sY = sX + 128 * 32;              // 128*32 f16 (8 KB)
    float*    sL = (float*)smem;               // 128*128 f32 after GEMM

    int tid = threadIdx.x, lane = tid & 31, wv = tid >> 5;
    v8f acc[8];
    #pragma unroll
    for (int i = 0; i < 8; i++)
        #pragma unroll
        for (int r = 0; r < 8; r++) acc[i][r] = 0.0f;

    for (int kc = 0; kc < 8; kc++) {
        __syncthreads();
        for (int i = tid; i < 128 * 32; i += 256) {
            int a = i >> 5, k = i & 31;
            sX[i] = x1[a * KCH + kc * 32 + k];
            sY[i] = y1[a * KCH + kc * 32 + k];
        }
        __syncthreads();
        v16h afrag;
        int khi = (lane >> 4) * 8;
        {
            int prow = wv * 16 + (lane & 15);
            #pragma unroll
            for (int e = 0; e < 16; e++) {
                int k = (e < 8 ? e : e + 8) + khi;
                afrag[e] = sX[prow * 32 + k];
            }
        }
        #pragma unroll
        for (int nt = 0; nt < 8; nt++) {
            v16h bfrag;
            int ncol = nt * 16 + (lane & 15);
            #pragma unroll
            for (int e = 0; e < 16; e++) {
                int k = (e < 8 ? e : e + 8) + khi;
                bfrag[e] = sY[ncol * 32 + k];
            }
            acc[nt] = __builtin_amdgcn_wmma_f32_16x16x32_f16(
                false, afrag, false, bfrag, (short)0, acc[nt], false, false);
        }
    }
    __syncthreads();
    const float inv_temp = 1.0f / 25.0f;
    int mrow0 = (lane >> 4) * 8;
    int ncl   = lane & 15;
    #pragma unroll
    for (int nt = 0; nt < 8; nt++)
        #pragma unroll
        for (int r = 0; r < 8; r++)
            sL[(wv * 16 + mrow0 + r) * 128 + nt * 16 + ncl] = acc[nt][r] * inv_temp;
    __syncthreads();

    float contrib = 0.0f;
    if (tid < 128) {
        float mx = sL[tid * 128];
        for (int j = 1; j < 128; j++) mx = fmaxf(mx, sL[tid * 128 + j]);
        float se = 0.0f;
        for (int j = 0; j < 128; j++) se += expf(sL[tid * 128 + j] - mx);
        contrib = (mx + logf(se)) - sL[tid * 128 + tid];   // -log_softmax diag
    }
    __syncthreads();
    float* sRed = (float*)smem;      // reuse
    sRed[tid] = contrib;
    __syncthreads();
    for (int s = 128; s > 0; s >>= 1) {
        if (tid < s) sRed[tid] += sRed[tid + s];
        __syncthreads();
    }
    if (tid == 0) acc_g[1] = sRed[0] / 128.0f;
}

// ---------------------------------------------------------------------------
// OHEM: smooth-L1 between reflection-sampled k1 and k2, per-frame sums.
__global__ __launch_bounds__(256)
void ohem_kernel(const float* __restrict__ k1, const float* __restrict__ k2,
                 const int* __restrict__ srcR, float* __restrict__ acc_g) {
    int bid = blockIdx.x;            // 32 frames * 32 chunks
    int n   = bid >> 5;
    int p0  = (bid & 31) * 128;
    int tid = threadIdx.x;
    int pl  = tid & 127;
    int c0  = tid >> 7;              // 0 or 1
    size_t fbase = (size_t)n * KCH * HWSZ;
    int s = srcR[n * HWSZ + p0 + pl];

    float sum = 0.0f;
    for (int c = c0; c < KCH; c += 2) {
        float a = k1[fbase + (size_t)c * HWSZ + s];
        float b = k2[fbase + (size_t)c * HWSZ + p0 + pl];
        float d = a - b;
        float ad = fabsf(d);
        sum += (ad < 0.5f) ? d * d : (ad - 0.25f);   // beta=0.5: 0.5*d*d/beta = d*d
    }
    __shared__ float sRed[256];
    sRed[tid] = sum;
    __syncthreads();
    for (int st = 128; st > 0; st >>= 1) {
        if (tid < st) sRed[tid] += sRed[tid + st];
        __syncthreads();
    }
    if (tid == 0) atomicAdd(&acc_g[2 + n], sRed[0]);
}

// ---------------------------------------------------------------------------
__global__ void finalize_kernel(const float* __restrict__ acc_g, float* __restrict__ out) {
    if (threadIdx.x == 0) {
        float pf[BTF];
        for (int i = 0; i < BTF; i++) pf[i] = acc_g[2 + i] / (float)(KCH * HWSZ);
        // insertion sort ascending
        for (int i = 1; i < BTF; i++) {
            float v = pf[i]; int j = i - 1;
            while (j >= 0 && pf[j] > v) { pf[j + 1] = pf[j]; j--; }
            pf[j + 1] = v;
        }
        float oh = 0.0f;
        for (int i = 12; i < 32; i++) oh += pf[i];     // i1=int(32*0.4)=12, i2=32
        oh *= (1.0f / 20.0f);
        float lt = acc_g[0] / (float)(8 * 3 * HWSZ);   // B*(T-1)*H*W
        out[0] = lt + oh + 0.1f * acc_g[1];
    }
}

// ---------------------------------------------------------------------------
extern "C" void kernel_launch(void* const* d_in, const int* in_sizes, int n_in,
                              void* d_out, int out_size, void* d_ws, size_t ws_size,
                              hipStream_t stream) {
    const float* k1      = (const float*)d_in[0];
    const float* k2      = (const float*)d_in[1];
    const float* aff1    = (const float*)d_in[2];
    const float* aff2    = (const float*)d_in[3];
    const float* affgrid = (const float*)d_in[4];
    const int*   hw_idx  = (const int*)d_in[5];
    const int*   hw_ref  = (const int*)d_in[6];

    unsigned char* ws = (unsigned char*)d_ws;
    int*      src1 = (int*)(ws + OFF_SRC1);
    int*      src2 = (int*)(ws + OFF_SRC2);
    int*      srcR = (int*)(ws + OFF_SRCR);
    _Float16* q1   = (_Float16*)(ws + OFF_Q1);
    _Float16* x1v  = (_Float16*)(ws + OFF_X1);
    _Float16* y1v  = (_Float16*)(ws + OFF_Y1);
    float*    accs = (float*)(ws + OFF_ACC);

    init_kernel<<<1, 64, 0, stream>>>(accs);
    precompute_kernel<<<dim3(16, 32), 256, 0, stream>>>(aff1, aff2, affgrid, src1, src2, srcR);
    anchors_kernel<<<128, 256, 0, stream>>>(k1, hw_idx, q1);
    refgather_kernel<<<128, 256, 0, stream>>>(k1, k2, src1, src2, hw_ref, x1v, y1v);
    temp_loss_kernel<<<24 * 64, 256, 0, stream>>>(k1, k2, src1, src2, q1, accs);
    ref_loss_kernel<<<1, 256, 0, stream>>>(x1v, y1v, accs);
    ohem_kernel<<<32 * 32, 256, 0, stream>>>(k1, k2, srcR, accs);
    finalize_kernel<<<1, 32, 0, stream>>>(accs, (float*)d_out);
}
